// Long_term_atention_9113920602538
// MI455X (gfx1250) — matrix operand order repaired and checked by reading, not compiled, not run on hardware
//
#include <hip/hip_runtime.h>
#include <hip/hip_bf16.h>
#include <math.h>

// Problem constants
#define B_  16
#define KL_ 2048
#define D_  512
#define ROWS_ (B_ * KL_)          // 32768
#define LN_EPS_ 1e-5f

typedef __attribute__((ext_vector_type(16))) __bf16 v16bf;
typedef __attribute__((ext_vector_type(8)))  float  v8f;

// ---------------------------------------------------------------------------
// 1) Q_ = Q @ W_q   (16x512) -- tiny fp32
__global__ void k_proj_q(const float* __restrict__ Q, const float* __restrict__ Wq,
                         float* __restrict__ Qp) {
    int i = blockIdx.x * blockDim.x + threadIdx.x;      // 8192
    int b = i >> 9, d = i & 511;
    float acc = 0.f;
    for (int e = 0; e < D_; ++e) acc += Q[b * D_ + e] * Wq[e * D_ + d];
    Qp[i] = acc;
}

// 2) qt[b,e] = (sum_d W_k[e,d] * Q_[b,d]) / temperature   (folds temp into scores)
__global__ void k_proj_qt(const float* __restrict__ Qp, const float* __restrict__ Wk,
                          float* __restrict__ qt) {
    int i = blockIdx.x * blockDim.x + threadIdx.x;      // 8192
    int b = i >> 9, e = i & 511;
    float acc = 0.f;
    for (int d = 0; d < D_; ++d) acc += Qp[b * D_ + d] * Wk[e * D_ + d];
    float invT = 1.0f / (sqrtf((float)D_) + 1e-6f);
    qt[i] = acc * invT;
}

// 3) Vcol[b,d] = sum_k V[b,k,d]   (for uniform-attention masked rows)
__global__ void k_vcol(const float* __restrict__ V, float* __restrict__ Vcol) {
    int i = blockIdx.x * blockDim.x + threadIdx.x;      // 8192
    int b = i >> 9, d = i & 511;
    const float* p = V + (size_t)b * KL_ * D_ + d;
    float acc = 0.f;
    for (int k = 0; k < KL_; ++k) acc += p[(size_t)k * D_];
    Vcol[i] = acc;
}

// 4) Vsum[b,n] = (Vcol[b,:] @ W_v)[n] / K_LEN  == mean_k V_[b,k,n]
__global__ void k_vsum_proj(const float* __restrict__ Vcol, const float* __restrict__ Wv,
                            float* __restrict__ Vsum) {
    int i = blockIdx.x * blockDim.x + threadIdx.x;      // 8192
    int b = i >> 9, n = i & 511;
    float acc = 0.f;
    for (int e = 0; e < D_; ++e) acc += Vcol[b * D_ + e] * Wv[e * D_ + n];
    Vsum[i] = acc * (1.0f / (float)KL_);
}

// 5) s[b,k] = K[b,k,:] . qt[b,:]   -- one wave32 per row, shuffle reduction
__global__ void k_scores(const float* __restrict__ K, const float* __restrict__ qt,
                         float* __restrict__ s) {
    int wid  = (blockIdx.x * blockDim.x + threadIdx.x) >> 5;   // 0..32767
    int lane = threadIdx.x & 31;
    int b = wid >> 11;
    const float* Kr = K + (size_t)wid * D_;
    const float* qb = qt + b * D_;
    float acc = 0.f;
#pragma unroll
    for (int j = 0; j < 16; ++j) { int d = lane + 32 * j; acc += Kr[d] * qb[d]; }
#pragma unroll
    for (int off = 16; off > 0; off >>= 1) acc += __shfl_xor(acc, off, 32);
    if (lane == 0) s[wid] = acc;
}

// 6) Per batch: M = max(s), P = exp(s-M), rden = 1 / inclusive-prefix-sum(P)
__global__ void k_softmax_prefix(const float* __restrict__ s, float* __restrict__ P,
                                 float* __restrict__ rden) {
    __shared__ float buf[KL_];
    __shared__ float part[256];
    int b = blockIdx.x, t = threadIdx.x;
    const float* sb = s + b * KL_;
#pragma unroll
    for (int j = 0; j < 8; ++j) buf[t + 256 * j] = sb[t + 256 * j];
    __syncthreads();
    float m = -3.4e38f;
#pragma unroll
    for (int e = 0; e < 8; ++e) m = fmaxf(m, buf[t * 8 + e]);
    part[t] = m; __syncthreads();
    for (int st = 128; st > 0; st >>= 1) {
        if (t < st) part[t] = fmaxf(part[t], part[t + st]);
        __syncthreads();
    }
    float M = part[0]; __syncthreads();
    float loc = 0.f;
#pragma unroll
    for (int e = 0; e < 8; ++e) {
        float p = __expf(buf[t * 8 + e] - M);
        buf[t * 8 + e] = p; loc += p;
    }
    part[t] = loc; __syncthreads();
    if (t == 0) {                        // serial scan of 256 partials (cheap)
        float run = 0.f;
        for (int i = 0; i < 256; ++i) { float tmp = part[i]; part[i] = run; run += tmp; }
    }
    __syncthreads();
    float run = part[t];
#pragma unroll
    for (int e = 0; e < 8; ++e) {
        float p = buf[t * 8 + e];
        run += p;
        rden[b * KL_ + t * 8 + e] = 1.0f / run;   // reciprocal: scan loop multiplies
        P[b * KL_ + t * 8 + e]   = p;
    }
}

// 7a) Pre-transpose + convert W_v to bf16:  Wt[n][k] = bf16(W_v[k][n])
//     (512 KB, L2-resident; gives contiguous per-lane B-fragment loads)
__global__ void k_wcvt(const float* __restrict__ W, __bf16* __restrict__ Wt) {
    int i = blockIdx.x * blockDim.x + threadIdx.x;   // 262144
    int n = i >> 9, k = i & 511;
    Wt[(size_t)n * D_ + k] = (__bf16)W[(size_t)k * D_ + n];
}

// 7b) V_ = V @ W_v via v_wmma_f32_16x16x32_bf16.
//     One wave computes a 16x64 output strip: A fragment reused across 4 WMMAs,
//     B fragments are single contiguous 32B bf16 loads per lane from Wt.
//     Writes V_ into `out` (consumed and overwritten in place by the scan).
__global__ void k_gemm_v(const float* __restrict__ V, const __bf16* __restrict__ Wt,
                         float* __restrict__ out) {
    const int lane = threadIdx.x & 31;
    const int wave = threadIdx.x >> 5;
    const int gid  = blockIdx.x * 8 + wave;          // 16384 wave-strips
    const int tm   = gid >> 3;                       // 2048 M-tiles
    const int tn4  = gid & 7;                        // 8 N-groups of 64
    const int row0 = tm * 16;
    const int col0 = tn4 * 64;
    const int half = lane >> 4;                      // 0/1
    const int l15  = lane & 15;

    const float* Arow = V + (size_t)(row0 + l15) * D_;
    const __bf16* B0 = Wt + (size_t)(col0 +  0 + l15) * D_;
    const __bf16* B1 = Wt + (size_t)(col0 + 16 + l15) * D_;
    const __bf16* B2 = Wt + (size_t)(col0 + 32 + l15) * D_;
    const __bf16* B3 = Wt + (size_t)(col0 + 48 + l15) * D_;

    v8f acc0 = {}, acc1 = {}, acc2 = {}, acc3 = {};
#pragma unroll 2
    for (int kk = 0; kk < D_; kk += 32) {
        // A (16x32 bf16): lane<16 holds K {0..15}, lane>=16 holds K {16..31}
        const float4* ap0 = (const float4*)(Arow + kk + half * 16);
        const float4* ap1 = (const float4*)(Arow + kk + half * 16 + 8);
        float4 f0 = ap0[0], f1 = ap0[1], f2 = ap1[0], f3 = ap1[1];
        v16bf a;
        a[0]=(__bf16)f0.x; a[1]=(__bf16)f0.y; a[2]=(__bf16)f0.z; a[3]=(__bf16)f0.w;
        a[4]=(__bf16)f1.x; a[5]=(__bf16)f1.y; a[6]=(__bf16)f1.z; a[7]=(__bf16)f1.w;
        a[8]=(__bf16)f2.x; a[9]=(__bf16)f2.y; a[10]=(__bf16)f2.z; a[11]=(__bf16)f2.w;
        a[12]=(__bf16)f3.x; a[13]=(__bf16)f3.y; a[14]=(__bf16)f3.z; a[15]=(__bf16)f3.w;
        // B (32x16 bf16): lane = column N; halves split K 0..15 / 16..31
        const int kb = kk + half * 16;               // 32B-aligned offset in Wt rows
        v16bf b0 = *(const v16bf*)(B0 + kb);
        v16bf b1 = *(const v16bf*)(B1 + kb);
        v16bf b2 = *(const v16bf*)(B2 + kb);
        v16bf b3 = *(const v16bf*)(B3 + kb);
        acc0 = __builtin_amdgcn_wmma_f32_16x16x32_bf16(false, a, false, b0, (short)0, acc0, false, false);
        acc1 = __builtin_amdgcn_wmma_f32_16x16x32_bf16(false, a, false, b1, (short)0, acc1, false, false);
        acc2 = __builtin_amdgcn_wmma_f32_16x16x32_bf16(false, a, false, b2, (short)0, acc2, false, false);
        acc3 = __builtin_amdgcn_wmma_f32_16x16x32_bf16(false, a, false, b3, (short)0, acc3, false, false);
    }
    // D (16x16 f32): VGPR r -> row = row0 + half*8 + r, col = col0 + 16*t + l15
    float* Orow = out + (size_t)(row0 + half * 8) * D_ + col0 + l15;
#pragma unroll
    for (int r = 0; r < 8; ++r) {
        Orow[(size_t)r * D_ +  0] = acc0[r];
        Orow[(size_t)r * D_ + 16] = acc1[r];
        Orow[(size_t)r * D_ + 32] = acc2[r];
        Orow[(size_t)r * D_ + 48] = acc3[r];
    }
}

// 8) Sequential scan over q: out[b,q,d] = V[b,q,d] + att
//    att = mask[b,q] ? mean_k V_[b,k,d] : prefix(P*V_) * rden[q]
//    out currently holds V_; each element read once, then overwritten.
__global__ void k_scan(const float* __restrict__ V, const unsigned char* __restrict__ mask,
                       const float* __restrict__ P, const float* __restrict__ rden,
                       const float* __restrict__ Vsum, float* __restrict__ out) {
    int b = blockIdx.x >> 1;
    int d = ((blockIdx.x & 1) << 8) + threadIdx.x;    // 0..511
    const float* Pb = P + b * KL_;
    const float* Rb = rden + b * KL_;
    const unsigned char* Mb = mask + b * KL_;
    float vs = Vsum[b * D_ + d];
    size_t base = (size_t)b * KL_ * D_ + d;
    float acc = 0.f;
    for (int q = 0; q < KL_; ++q) {
        size_t idx = base + (size_t)q * D_;
        float vproj = out[idx];                  // V_ row element
        acc += Pb[q] * vproj;
        float att = Mb[q] ? vs : (acc * Rb[q]);
        out[idx] = V[idx] + att;
    }
}

// 9) In-place LayerNorm, one wave32 per 512-elem row
__global__ void k_layernorm(float* __restrict__ out, const float* __restrict__ g,
                            const float* __restrict__ be) {
    int wid  = (blockIdx.x * blockDim.x + threadIdx.x) >> 5;  // 0..32767
    int lane = threadIdx.x & 31;
    float* row = out + (size_t)wid * D_;
    float x[16];
    float sum = 0.f;
#pragma unroll
    for (int j = 0; j < 16; ++j) { x[j] = row[lane + 32 * j]; sum += x[j]; }
#pragma unroll
    for (int off = 16; off > 0; off >>= 1) sum += __shfl_xor(sum, off, 32);
    float mu = sum * (1.0f / (float)D_);
    float vs = 0.f;
#pragma unroll
    for (int j = 0; j < 16; ++j) { float t = x[j] - mu; vs += t * t; }
#pragma unroll
    for (int off = 16; off > 0; off >>= 1) vs += __shfl_xor(vs, off, 32);
    float inv = rsqrtf(vs * (1.0f / (float)D_) + LN_EPS_);
#pragma unroll
    for (int j = 0; j < 16; ++j) {
        int d = lane + 32 * j;
        row[d] = (x[j] - mu) * inv * g[d] + be[d];
    }
}

// ---------------------------------------------------------------------------
extern "C" void kernel_launch(void* const* d_in, const int* in_sizes, int n_in,
                              void* d_out, int out_size, void* d_ws, size_t ws_size,
                              hipStream_t stream) {
    const float* Q  = (const float*)d_in[0];
    const float* K  = (const float*)d_in[1];
    const float* V  = (const float*)d_in[2];
    const unsigned char* mask = (const unsigned char*)d_in[3];
    const float* Wq = (const float*)d_in[4];
    const float* Wk = (const float*)d_in[5];
    const float* Wv = (const float*)d_in[6];
    const float* g  = (const float*)d_in[7];
    const float* be = (const float*)d_in[8];
    float* out = (float*)d_out;

    float* ws   = (float*)d_ws;
    float* Qp   = ws;                 // 8192
    float* qt   = ws + 8192;          // 8192
    float* s    = ws + 16384;         // 32768
    float* P    = ws + 49152;         // 32768
    float* rden = ws + 81920;         // 32768
    float* Vcol = ws + 114688;        // 8192
    float* Vsum = ws + 122880;        // 8192
    __bf16* Wt  = (__bf16*)(ws + 131072);   // 262144 bf16 = 512 KB (total 1 MB ws)

    k_proj_q        <<<32,   256, 0, stream>>>(Q, Wq, Qp);
    k_proj_qt       <<<32,   256, 0, stream>>>(Qp, Wk, qt);
    k_vcol          <<<32,   256, 0, stream>>>(V, Vcol);
    k_vsum_proj     <<<32,   256, 0, stream>>>(Vcol, Wv, Vsum);
    k_scores        <<<4096, 256, 0, stream>>>(K, qt, s);
    k_softmax_prefix<<<B_,   256, 0, stream>>>(s, P, rden);
    k_wcvt          <<<1024, 256, 0, stream>>>(Wv, Wt);
    k_gemm_v        <<<2048, 256, 0, stream>>>(V, Wt, out);    // V_ -> out
    k_scan          <<<32,   256, 0, stream>>>(V, mask, P, rden, Vsum, out);
    k_layernorm     <<<4096, 256, 0, stream>>>(out, g, be);
}
